// MultiHeadedAttention_88613765251147
// MI455X (gfx1250) — matrix operand order repaired and checked by reading, not compiled, not run on hardware
//
#include <hip/hip_runtime.h>
#include <hip/hip_bf16.h>

// ---------------------------------------------------------------------------
// MI455X (gfx1250) multi-head attention pipeline.
// All matmuls: v_wmma_f32_16x16x32_f16 (f32 accumulation).
// A and B GEMM tiles + attention K tiles staged with
// global_load_async_to_lds_b128 (saddr form, ASYNCcnt) straight into
// fragment-ordered, double-buffered LDS.
// ---------------------------------------------------------------------------

typedef __attribute__((ext_vector_type(16))) _Float16 v16h;
typedef __attribute__((ext_vector_type(8)))  float    v8f;

#define B_     2
#define S_     1024
#define HID_   2048
#define H_     32
#define D_     64
#define KV_    8
#define NREP_  4
#define INNER_ 2048
#define KVDIM_ 512

#define LDS_AS __attribute__((address_space(3)))

__device__ __forceinline__ unsigned pack2h(float a, float b) {
  _Float16 ha = (_Float16)a, hb = (_Float16)b;
  unsigned short ua = __builtin_bit_cast(unsigned short, ha);
  unsigned short ub = __builtin_bit_cast(unsigned short, hb);
  return (unsigned)ua | ((unsigned)ub << 16);
}

__device__ __forceinline__ v8f wmma_f16(v16h a, v16h b, v8f c) {
  return __builtin_amdgcn_wmma_f32_16x16x32_f16(false, a, false, b,
                                                (short)0, c, false, false);
}

__device__ __forceinline__ unsigned lds_off_u32(const void* p) {
  return (unsigned)(size_t)(LDS_AS const void*)p;
}

// Async 16B global->LDS copy, GVS addressing: mem = saddr + voff(bytes).
__device__ __forceinline__ void async_b128_s(const void* lds_dst,
                                             const void* sbase, unsigned voff) {
  unsigned loff = lds_off_u32(lds_dst);
  asm volatile("global_load_async_to_lds_b128 %0, %1, %2"
               :: "v"(loff), "v"(voff), "s"(sbase) : "memory");
}
__device__ __forceinline__ void wait_async0() {
  asm volatile("s_wait_asynccnt 0x0" ::: "memory");
}

// A-fragment (16x32 f16): lane = ((k%16)/8)*16 + m ; elem i = (k/16)*8 + k%8
//   -> per lane two contiguous 16B runs: k = hh*8.. , 16+hh*8..
// B-fragment (32x16 f16): lane = ((k%16)/8)*16 + n ; elem i = (k/16)*8 + k%8
//   -> per lane two contiguous 16B runs, same k order as A.

// ---------------------------------------------------------------------------
// Weight transpose+convert: W[K,N] f32 -> Wt[N,K] f16.  64x64 LDS tile.
// ---------------------------------------------------------------------------
__global__ __launch_bounds__(256) void wtrans_kernel(
    const float* __restrict__ W, _Float16* __restrict__ Wt, int K, int N)
{
  __shared__ __align__(16) _Float16 tile[64 * 72];   // [n][k], stride 72
  const int t = threadIdx.x;
  const int n0 = blockIdx.x * 64, k0 = blockIdx.y * 64;
  {
    int kr = t >> 2;
    int c0 = 16 * (t & 3);
    const float* src = W + (size_t)(k0 + kr) * N + n0 + c0;
    #pragma unroll
    for (int q4 = 0; q4 < 4; ++q4) {
      float4 v = *(const float4*)(src + 4 * q4);
      float vv[4] = {v.x, v.y, v.z, v.w};
      #pragma unroll
      for (int c = 0; c < 4; ++c)
        tile[(c0 + 4 * q4 + c) * 72 + kr] = (_Float16)vv[c];
    }
  }
  __syncthreads();
  {
    int n = t >> 2;
    int kc0 = 16 * (t & 3);
    const uint4* src = (const uint4*)(tile + n * 72 + kc0);
    uint4* dst = (uint4*)(Wt + (size_t)(n0 + n) * K + k0 + kc0);
    dst[0] = src[0];
    dst[1] = src[1];
  }
}

// ---------------------------------------------------------------------------
// Elementwise f32 -> f16 (vectorized by 4).
// ---------------------------------------------------------------------------
__global__ __launch_bounds__(256) void cvt16_kernel(
    const float* __restrict__ in, _Float16* __restrict__ out, int n4)
{
  int i = blockIdx.x * 256 + threadIdx.x;
  if (i >= n4) return;
  float4 v = ((const float4*)in)[i];
  uint2 u;
  u.x = pack2h(v.x, v.y);
  u.y = pack2h(v.z, v.w);
  ((uint2*)out)[i] = u;
}

// ---------------------------------------------------------------------------
// GEMM: C[M,N] f32 = A16[M,K] @ Wt16[N,K]^T + bias.
// 64x64 tile, K-step 64, double-buffered LDS; BOTH A and B staged with
// async b128 (4 per thread per step).  8 waves, 4 WMMAs per wave per step.
// ---------------------------------------------------------------------------
__global__ __launch_bounds__(256) void gemm_f16_kernel(
    const _Float16* __restrict__ A16, const _Float16* __restrict__ Wt16,
    const float* __restrict__ bias, float* __restrict__ C,
    int M, int N, int K)
{
  __shared__ __align__(32) _Float16 shA[2][4096];   // [rg4][ks2][lane][16h]
  __shared__ __align__(32) _Float16 shB[2][4096];   // [cg4][ks2][lane][16h]
  const int t = threadIdx.x, lane = t & 31, w = t >> 5;
  const int wm = w & 3, wn = w >> 2;
  const int m0 = blockIdx.y * 64, n0 = blockIdx.x * 64;
  v8f acc0 = {}, acc1 = {};

  // A staging: thread -> fragment lane position (rg, ks, ln); 2 chunks (g)
  const int s_rg = t >> 6, s_ks = (t >> 5) & 1, s_ln = t & 31;
  const int s_m = s_ln & 15, s_hh = s_ln >> 4;
  const unsigned a_voff =
      (unsigned)((((size_t)(m0 + s_rg * 16 + s_m)) * K + s_ks * 32 + s_hh * 8) * 2);
  // B staging: thread -> Wt row bn, chunks bc0 and bc0+4
  const int bn = t >> 2, bc0 = t & 3;
  const int b_cg = bn >> 4, b_nn = bn & 15;

  auto stage = [&](int buf, int k0) {
    _Float16* dA = shA[buf] + ((s_rg * 2 + s_ks) * 32 + s_ln) * 16;
    const _Float16* abase = A16 + k0;
    async_b128_s(dA,     abase, a_voff);
    async_b128_s(dA + 8, abase, a_voff + 32);
    const _Float16* bbase = Wt16 + k0;
    #pragma unroll
    for (int cc = bc0; cc < 8; cc += 4) {
      int ks = cc >> 2;
      int ln = (cc & 1) * 16 + b_nn;
      unsigned voff = (unsigned)((((size_t)(n0 + bn)) * K + 8 * cc) * 2);
      async_b128_s(&shB[buf][((b_cg * 2 + ks) * 32 + ln) * 16 + ((cc >> 1) & 1) * 8],
                   bbase, voff);
    }
  };

  auto compute = [&](int buf) {
    const _Float16* fA = shA[buf];
    const _Float16* fB = shB[buf];
    v16h a0  = *(const v16h*)(fA + ((wm * 2 + 0) * 32 + lane) * 16);
    v16h a1  = *(const v16h*)(fA + ((wm * 2 + 1) * 32 + lane) * 16);
    v16h b00 = *(const v16h*)(fB + (((wn * 2 + 0) * 2 + 0) * 32 + lane) * 16);
    v16h b01 = *(const v16h*)(fB + (((wn * 2 + 0) * 2 + 1) * 32 + lane) * 16);
    v16h b10 = *(const v16h*)(fB + (((wn * 2 + 1) * 2 + 0) * 32 + lane) * 16);
    v16h b11 = *(const v16h*)(fB + (((wn * 2 + 1) * 2 + 1) * 32 + lane) * 16);
    acc0 = wmma_f16(a0, b00, acc0);
    acc0 = wmma_f16(a1, b01, acc0);
    acc1 = wmma_f16(a0, b10, acc1);
    acc1 = wmma_f16(a1, b11, acc1);
  };

  stage(0, 0);
  wait_async0();
  __syncthreads();
  const int nIter = K >> 6;
  for (int it = 0; it < nIter; ++it) {
    int buf = it & 1;
    if (it + 1 < nIter) stage(buf ^ 1, (it + 1) << 6);
    compute(buf);
    wait_async0();
    __syncthreads();
  }

  const int row_base = m0 + wm * 16 + ((lane >> 4) << 3);
  const int col0 = n0 + wn * 32 + (lane & 15);
  const float bias0 = bias[col0];
  const float bias1 = bias[col0 + 16];
  #pragma unroll
  for (int j = 0; j < 8; ++j) {
    int row = row_base + j;
    C[(size_t)row * N + col0]      = acc0[j] + bias0;
    C[(size_t)row * N + col0 + 16] = acc1[j] + bias1;
  }
}

// ---------------------------------------------------------------------------
// Interleaved-pair RoPE: reads fp32 (B,S,nheads,D), writes f16 copy.
// ---------------------------------------------------------------------------
__global__ __launch_bounds__(256) void rope16_kernel(
    const float* __restrict__ x, _Float16* __restrict__ out,
    const float* __restrict__ fcos, const float* __restrict__ fsin,
    int nheads, int total_pairs)
{
  int n = blockIdx.x * blockDim.x + threadIdx.x;
  if (n >= total_pairs) return;
  int d2 = n & 31;
  int rest = n >> 5;
  int s = (rest / nheads) % S_;
  float c  = fcos[s * 32 + d2];
  float sn = fsin[s * 32 + d2];
  size_t base = (size_t)rest * 64 + 2 * d2;
  float x0 = x[base], x1 = x[base + 1];
  ((unsigned*)out)[base >> 1] = pack2h(x0 * c - x1 * sn, x0 * sn + x1 * c);
}

// ---------------------------------------------------------------------------
// scores = (q.k^T + pos_bias)/8 + mask + prev_attn -> prev_attn_out
// q16/k16 are f16; q staged with dword copies, K tiles with async b128.
// ---------------------------------------------------------------------------
__global__ __launch_bounds__(256) void scores_kernel(
    const _Float16* __restrict__ q16, const _Float16* __restrict__ k16,
    const float* __restrict__ pos_bias, const float* __restrict__ mask,
    const float* __restrict__ prev_attn, float* __restrict__ scores)
{
  __shared__ __align__(32) unsigned  shQ[1024];  // 32x64 A-frag
  __shared__ __align__(32) _Float16  shK[4096];  // 64x64 B-frag
  const int t = threadIdx.x, lane = t & 31, w = t >> 5;
  const int b = blockIdx.y / H_, h = blockIdx.y % H_;
  const int kvh = h / NREP_;
  const int q0 = blockIdx.x * 32;
  const _Float16* fQ = (const _Float16*)shQ;
  const _Float16* fK = shK;

  { // stage q tile (32 rows x 64 d) once: pure dword copies
    int r = t >> 3, dw0 = 4 * (t & 7);
    const uint4 u = *(const uint4*)(q16 + ((size_t)(b * S_ + q0 + r) * H_ + h) * D_ + 2 * dw0);
    unsigned dv[4] = {u.x, u.y, u.z, u.w};
    int rg = r >> 4, m = r & 15;
    #pragma unroll
    for (int c = 0; c < 4; ++c) {
      int kk = 2 * (dw0 + c);
      int ks = kk >> 5, ww = kk & 31;
      int hh = (ww & 15) >> 3, g = ww >> 4;
      int i  = g * 8 + (ww & 7);
      shQ[((rg * 2 + ks) * 32 + hh * 16 + m) * 8 + (i >> 1)] = dv[c];
    }
  }
  __syncthreads();
  const int wq = w & 1, wk = w >> 1;
  const v16h aq0 = *(const v16h*)(fQ + ((wq * 2 + 0) * 32 + lane) * 16);
  const v16h aq1 = *(const v16h*)(fQ + ((wq * 2 + 1) * 32 + lane) * 16);

  // K-tile async staging: thread -> (row kr, chunk: ks, half-sel)
  const int kr = t >> 2, kcc = t & 3, kks = kcc >> 1, khs = kcc & 1;
  const unsigned k_voff = (unsigned)(((size_t)kr * KV_ * D_ + kks * 32 + khs * 8) * 2);
  _Float16* kdst = shK + (((kr >> 4) * 2 + kks) * 32 + khs * 16 + (kr & 15)) * 16;

  for (int kt = 0; kt < S_ / 64; ++kt) {
    __syncthreads();
    const _Float16* kbase = k16 + ((size_t)(b * S_ + kt * 64) * KV_ + kvh) * D_;
    async_b128_s(kdst,     kbase, k_voff);
    async_b128_s(kdst + 8, kbase, k_voff + 32);
    wait_async0();
    __syncthreads();
    v16h b0 = *(const v16h*)(fK + ((wk * 2 + 0) * 32 + lane) * 16);
    v16h b1 = *(const v16h*)(fK + ((wk * 2 + 1) * 32 + lane) * 16);
    v8f acc = {};
    acc = wmma_f16(aq0, b0, acc);
    acc = wmma_f16(aq1, b1, acc);
    const int qrow_base = q0 + wq * 16 + ((lane >> 4) << 3);
    const int kcol = kt * 64 + wk * 16 + (lane & 15);
    #pragma unroll
    for (int j = 0; j < 8; ++j) {
      int qrow = qrow_base + j;
      size_t idx = ((size_t)(b * H_ + h) * S_ + qrow) * S_ + kcol;
      float sv = (acc[j] + pos_bias[((size_t)h * S_ + qrow) * S_ + kcol]) * 0.125f
               + mask[((size_t)b * S_ + qrow) * S_ + kcol]
               + prev_attn[idx];
      scores[idx] = sv;
    }
  }
}

// ---------------------------------------------------------------------------
// softmax + ctx = probs @ V (WMMA).  Two passes over scores (max; exp+sum
// fused with P staging), 1/rowsum folded into the f16 ctx epilogue.
// ---------------------------------------------------------------------------
__global__ __launch_bounds__(256) void softmax_ctx_kernel(
    const float* __restrict__ scores, const float* __restrict__ v,
    _Float16* __restrict__ ctx16)
{
  __shared__ __align__(32) unsigned shP[8192];  // [rg2][ks16][lane][8dw] 32KB
  __shared__ __align__(32) unsigned shV[4096];  // [cg4][ks4][lane][8dw] 16KB
  __shared__ float red[256];
  __shared__ float rowmax[32];
  __shared__ float rinv[32];
  const _Float16* fP = (const _Float16*)shP;
  const _Float16* fV = (const _Float16*)shV;

  const int t = threadIdx.x, lane = t & 31, w = t >> 5;
  const int b = blockIdx.y / H_, h = blockIdx.y % H_;
  const int kvh = h / NREP_;
  const int q0 = blockIdx.x * 32;
  const int r  = t >> 3;
  const int c0 = t & 7;

  const float* srow = scores + ((size_t)(b * H_ + h) * S_ + q0 + r) * S_;

  // pass 1: row max
  float mx = -3.0e38f;
  for (int c = c0; c < S_; c += 8) mx = fmaxf(mx, srow[c]);
  red[t] = mx;
  __syncthreads();
  if (c0 == 0) {
    float m2 = red[t];
    #pragma unroll
    for (int i = 1; i < 8; ++i) m2 = fmaxf(m2, red[t + i]);
    rowmax[r] = m2;
  }
  __syncthreads();

  // pass 2: exp + P staging + running sum, interleaved with P@V chunks
  const float rm = rowmax[r];
  const int prg = w & 1, cg = w >> 1;
  const int rg = r >> 4, m = r & 15;
  const int kp = t >> 4;
  const int d0 = 4 * (t & 15);
  float sum = 0.0f;
  v8f acc = {};
  for (int hp = 0; hp < 2; ++hp) {
    for (int c = hp * 512 + c0; c < hp * 512 + 512; c += 8) {
      float e = __expf(srow[c] - rm);
      sum += e;
      int cl = c - hp * 512;
      int ks = cl >> 5, ww = cl & 31;
      int hh = (ww & 15) >> 3, g = ww >> 4;
      int i  = g * 8 + (ww & 7);
      ((_Float16*)shP)[((rg * 16 + ks) * 32 + hh * 16 + m) * 16 + i] = (_Float16)e;
    }
    for (int vc = 0; vc < 4; ++vc) {       // 128-key V chunks
      #pragma unroll
      for (int it = 0; it < 4; ++it) {
        int klocal = 2 * (kp + 16 * it);
        size_t key = (size_t)hp * 512 + vc * 128 + klocal;
        const float* s0 = v + ((size_t)(b * S_ + key    ) * KV_ + kvh) * D_ + d0;
        const float* s1 = v + ((size_t)(b * S_ + key + 1) * KV_ + kvh) * D_ + d0;
        float4 v0 = *(const float4*)s0;
        float4 v1 = *(const float4*)s1;
        float a0[4] = {v0.x, v0.y, v0.z, v0.w};
        float a1[4] = {v1.x, v1.y, v1.z, v1.w};
        #pragma unroll
        for (int c = 0; c < 4; ++c) {
          int d = d0 + c;
          int vcg = d >> 4, n = d & 15;
          int ks = klocal >> 5, ww = klocal & 31;
          int ln = ((ww & 15) >> 3) * 16 + n;
          shV[((vcg * 4 + ks) * 32 + ln) * 8 + (((ww >> 4) * 8 + (ww & 7)) >> 1)] =
              pack2h(a0[c], a1[c]);
        }
      }
      __syncthreads();
      #pragma unroll
      for (int ks = 0; ks < 4; ++ks) {
        v16h a  = *(const v16h*)(fP + ((prg * 16 + vc * 4 + ks) * 32 + lane) * 16);
        v16h bb = *(const v16h*)(fV + ((cg * 4 + ks) * 32 + lane) * 16);
        acc = wmma_f16(a, bb, acc);
      }
      __syncthreads();
    }
  }
  // row-sum reduction (after P@V; normalization is linear)
  red[t] = sum;
  __syncthreads();
  if (c0 == 0) {
    float s2 = 0.0f;
    #pragma unroll
    for (int i = 0; i < 8; ++i) s2 += red[t + i];
    rinv[r] = 1.0f / s2;
  }
  __syncthreads();
  #pragma unroll
  for (int j = 0; j < 8; ++j) {
    int rl = prg * 16 + j + ((lane >> 4) << 3);
    int qpos = q0 + rl;
    int d = cg * 16 + (lane & 15);
    ctx16[((size_t)(b * S_ + qpos) * H_ + h) * D_ + d] = (_Float16)(acc[j] * rinv[rl]);
  }
}

// ---------------------------------------------------------------------------
extern "C" void kernel_launch(void* const* d_in, const int* in_sizes, int n_in,
                              void* d_out, int out_size, void* d_ws, size_t ws_size,
                              hipStream_t stream) {
  const float* query     = (const float*)d_in[0];
  const float* key       = (const float*)d_in[1];
  const float* value     = (const float*)d_in[2];
  const float* mask      = (const float*)d_in[3];
  const float* pos_bias  = (const float*)d_in[4];
  const float* prev_attn = (const float*)d_in[5];
  const float* fcos      = (const float*)d_in[6];
  const float* fsin      = (const float*)d_in[7];
  const float* Wq = (const float*)d_in[8];
  const float* bq = (const float*)d_in[9];
  const float* Wk = (const float*)d_in[10];
  const float* bk = (const float*)d_in[11];
  const float* Wv = (const float*)d_in[12];
  const float* bv = (const float*)d_in[13];
  const float* Wo = (const float*)d_in[14];
  const float* bo = (const float*)d_in[15];

  float* out        = (float*)d_out;
  float* scores_out = out + (size_t)B_ * S_ * HID_;

  const size_t ACT = (size_t)B_ * S_ * HID_;    // 4.19M
  const size_t KVA = (size_t)B_ * S_ * KVDIM_;  // 1.05M
  const size_t WQN = (size_t)HID_ * INNER_;
  const size_t WKN = (size_t)HID_ * KVDIM_;

  float* qbuf = (float*)d_ws;                   // (B,S,H,D)  fp32 (pre-RoPE)
  float* kbuf = qbuf + ACT;                     // (B,S,KV,D) fp32 (pre-RoPE)
  float* vbuf = kbuf + KVA;                     // (B,S,KV,D) fp32
  _Float16* hws = (_Float16*)(vbuf + KVA);
  _Float16* q16in = hws;  hws += ACT;
  _Float16* k16in = hws;  hws += ACT;
  _Float16* v16in = hws;  hws += ACT;
  _Float16* ctx16 = hws;  hws += ACT;
  _Float16* q16r  = hws;  hws += ACT;           // RoPE'd q, f16
  _Float16* k16r  = hws;  hws += KVA;           // RoPE'd k, f16
  _Float16* Wq16t = hws;  hws += WQN;
  _Float16* Wo16t = hws;  hws += WQN;
  _Float16* Wk16t = hws;  hws += WKN;
  _Float16* Wv16t = hws;  hws += WKN;

  dim3 blk(256);
  const int M = B_ * S_;

  // one-time weight transpose+convert (K x N f32 -> N x K f16)
  wtrans_kernel<<<dim3(INNER_ / 64, HID_ / 64), blk, 0, stream>>>(Wq, Wq16t, HID_, INNER_);
  wtrans_kernel<<<dim3(KVDIM_ / 64, HID_ / 64), blk, 0, stream>>>(Wk, Wk16t, HID_, KVDIM_);
  wtrans_kernel<<<dim3(KVDIM_ / 64, HID_ / 64), blk, 0, stream>>>(Wv, Wv16t, HID_, KVDIM_);
  wtrans_kernel<<<dim3(HID_ / 64, INNER_ / 64), blk, 0, stream>>>(Wo, Wo16t, INNER_, HID_);

  // activation converts
  int n4 = (int)(ACT / 4);
  cvt16_kernel<<<(n4 + 255) / 256, blk, 0, stream>>>(query, q16in, n4);
  cvt16_kernel<<<(n4 + 255) / 256, blk, 0, stream>>>(key,   k16in, n4);
  cvt16_kernel<<<(n4 + 255) / 256, blk, 0, stream>>>(value, v16in, n4);

  // projections (all-async double-buffered WMMA GEMM)
  gemm_f16_kernel<<<dim3(INNER_ / 64, M / 64), blk, 0, stream>>>(
      q16in, Wq16t, bq, qbuf, M, INNER_, HID_);
  gemm_f16_kernel<<<dim3(KVDIM_ / 64, M / 64), blk, 0, stream>>>(
      k16in, Wk16t, bk, kbuf, M, KVDIM_, HID_);
  gemm_f16_kernel<<<dim3(KVDIM_ / 64, M / 64), blk, 0, stream>>>(
      v16in, Wv16t, bv, vbuf, M, KVDIM_, HID_);

  // RoPE -> f16 buffers
  int qpairs = B_ * S_ * H_ * (D_ / 2);
  int kpairs = B_ * S_ * KV_ * (D_ / 2);
  rope16_kernel<<<(qpairs + 255) / 256, blk, 0, stream>>>(qbuf, q16r, fcos, fsin, H_, qpairs);
  rope16_kernel<<<(kpairs + 255) / 256, blk, 0, stream>>>(kbuf, k16r, fcos, fsin, KV_, kpairs);

  // scores -> prev_attn_out
  scores_kernel<<<dim3(S_ / 32, B_ * H_), blk, 0, stream>>>(
      q16r, k16r, pos_bias, mask, prev_attn, scores_out);

  // softmax + P@V -> ctx16
  softmax_ctx_kernel<<<dim3(S_ / 32, B_ * H_), blk, 0, stream>>>(
      scores_out, vbuf, ctx16);

  // output projection
  gemm_f16_kernel<<<dim3(HID_ / 64, M / 64), blk, 0, stream>>>(
      ctx16, Wo16t, bo, out, M, HID_, INNER_);
}
